// MSDeformAttnFuse_72679436583576
// MI455X (gfx1250) — compile-verified, matrix operand order checked by reading, not run on hardware
//
#include <hip/hip_runtime.h>

// ---------------------------------------------------------------------------
// MSDeformAttn fused pipeline for gfx1250 (MI455X), wave32 + WMMA bf16.
//   K0: pack_weights  one-time fp32 -> bf16 conversion into WMMA fragment
//                     order (so GEMM B-tiles are contiguous DMA copies)
//   K1: value GEMM   [B*Lq,256]x[256,256]  (A transposed-read from collab)
//   K2: proj GEMM    [B*Lq,256]x[256,96]   (W_off|W_attn fused, A from ego)
//   K3: gather       one wave per (b,m,q); lane = channel d (D=32 == wave32)
//   K4: out GEMM     [B*Lq,256]x[256,256]  + bias + residual, transposed store
//
// B tiles stream global->LDS via the Tensor Data Mover (TENSOR_LOAD_TO_LDS,
// TENSORcnt) when the builtin exists, else GLOBAL_LOAD_ASYNC_TO_LDS_B128
// (ASYNCcnt) via inline asm. A tiles are converted fp32->bf16 and scattered
// into fragment order; every lane's v16bf fragment is then one contiguous
// 32B LDS read (2x ds_load_b128) feeding back-to-back WMMA.
// ---------------------------------------------------------------------------

typedef __attribute__((ext_vector_type(16))) __bf16 v16bf;
typedef __attribute__((ext_vector_type(8)))  float  v8f;
typedef __attribute__((ext_vector_type(4)))  unsigned int u32x4;
typedef __attribute__((ext_vector_type(8)))  int i32x8;
typedef __attribute__((ext_vector_type(4)))  int i32x4;

#define B_  4
#define C_  256
#define H_  128
#define W_  128
#define M_  8
#define P_  4
#define D_  32
#define LQ  (H_ * W_)

#if defined(__has_builtin)
#if __has_builtin(__builtin_amdgcn_tensor_load_to_lds) && \
    __has_builtin(__builtin_amdgcn_s_wait_tensorcnt)
#define USE_TDM 1
#endif
#endif

__device__ inline __bf16 f2bf(float f) {
  unsigned u = __builtin_bit_cast(unsigned, f);
  unsigned r = u + 0x7FFFu + ((u >> 16) & 1u);   // round-to-nearest-even
  unsigned short h = (unsigned short)(r >> 16);
  return __builtin_bit_cast(__bf16, h);
}

// Inverse of the ISA 16-bit fragment layout (16x32 A tile / 32x16 B tile):
// element (k, m16) lives in lane = laneHalf*16 + m16 at vector slot `elem`.
__device__ inline void frag_slot(int k, int& laneHalf, int& elem) {
  laneHalf = (k >> 3) & 1;
  int vi   = ((k & 7) >> 1) + ((k >> 4) << 2);
  elem     = 2 * vi + (k & 1);
}

__device__ inline void async_b128_to_lds(unsigned lds_addr, const char* gaddr) {
  asm volatile("global_load_async_to_lds_b128 %0, %1, off"
               :: "v"(lds_addr), "v"(gaddr) : "memory");
}

// ---------------------------------------------------------------------------
// K0: convert a [256 x N] fp32 weight (optionally split Wa|Wb at splitN) into
// bf16 WMMA-fragment order: [kblock][ntile][lane][16 elems] (1KB per tile).
// ---------------------------------------------------------------------------
__global__ __launch_bounds__(256) void pack_weights_kernel(
    const float* __restrict__ Wa, const float* __restrict__ Wb,
    int N, int splitN, __bf16* __restrict__ outp) {
  int idx = blockIdx.x * 256 + threadIdx.x;   // over 256*N elements
  int k = idx / N;
  int n = idx % N;
  float v = (Wb != nullptr && n >= splitN)
                ? Wb[k * (N - splitN) + (n - splitN)]
                : Wa[k * splitN + n];
  int lh, el;
  frag_slot(k & 31, lh, el);
  int NTt = N >> 4;
  size_t o = (((size_t)(k >> 5) * NTt + (n >> 4)) * 32 + lh * 16 + (n & 15))
                 * 16 + el;
  outp[o] = f2bf(v);
}

// MODE 0: value GEMM   (A transposed from collab, bias=b_val)
// MODE 1: proj GEMM    (A transposed from ego, bias=b_off|b_attn)
// MODE 2: out GEMM     (A row-major, bias=b_out, bias2=ego residual,
//                       store transposed to [B, C, Lq])
template <int MODE, int BN>
__global__ __launch_bounds__(256) void gemm_wmma_kernel(
    const float* __restrict__ Asrc,
    const __bf16* __restrict__ Bpk,   // pre-packed fragment-order weights
    const float* __restrict__ bias,
    const float* __restrict__ bias2,
    float* __restrict__ Out) {
  constexpr int BM = 128;
  constexpr int BK = 32;
  constexpr int NT = BN / 16;                       // WMMA tiles along N
  constexpr int NTtot = (MODE == 1) ? 6 : 16;       // total N tiles in weight

  // Fragment-order LDS: [tile][lane][16 bf16] -> 32B contiguous per lane.
  __shared__ v16bf AfragLDS[BM / 16][32];
  __shared__ v16bf BfragLDS[NT][32];

  const int t        = threadIdx.x;
  const int wave     = t >> 5;
  const int lane     = t & 31;
  const int laneMod  = lane & 15;
  const int laneHalf = lane >> 4;

  const int rowBlock = blockIdx.x * BM;     // global row in [0, B*Lq)
  const int colBlock = blockIdx.y * BN;
  const int b  = rowBlock / LQ;             // BM divides LQ -> single image
  const int l0 = rowBlock % LQ;

  v8f zero = {};
  v8f acc[NT];
  #pragma unroll
  for (int i = 0; i < NT; ++i) acc[i] = zero;

  for (int k0 = 0; k0 < C_; k0 += BK) {
    // ------- B tile: one contiguous NT KB panel, global -> LDS -----------
    {
      const char* gbase = (const char*)(
          Bpk + ((((size_t)(k0 >> 5)) * NTtot + (colBlock >> 4)) << 9));
      char* lbase = (char*)&BfragLDS[0][0];
#if defined(USE_TDM)
      // Tensor Data Mover: single 1-D tile DMA per K-step (TENSORcnt).
      // D# per ISA ch.8: count=1, type=2("image"); data_size=3 (8B units);
      // tile_dim0 = tensor_dim0 = stride0 = NT*128 units; dims 1/2 trivial.
      if (wave == 0) {
        size_t   ga   = (size_t)gbase;
        unsigned lds  = (unsigned)(size_t)lbase;
        constexpr unsigned UNITS = (unsigned)(NT * 128);
        u32x4 g0 = {1u, lds, (unsigned)(ga & 0xFFFFFFFFu),
                    (unsigned)((ga >> 32) & 0x01FFFFFFu) | 0x80000000u};
        i32x8 g1 = {(int)(3u << 16),        // data_size=8B
                    (int)(UNITS << 16),     // tensor_dim0[15:0]
                    (int)(1u << 16),        // tensor_dim0[31:16]=0, tensor_dim1=1
                    (int)(UNITS << 16),     // tensor_dim1[31:16]=0, tile_dim0
                    1,                      // tile_dim1=1, tile_dim2=0
                    (int)UNITS,             // tensor_dim0_stride[31:0]
                    0, 0};
        i32x4 gz = {0, 0, 0, 0};
#if __clang_major__ >= 23
        i32x8 gz8 = {0, 0, 0, 0, 0, 0, 0, 0};
        __builtin_amdgcn_tensor_load_to_lds(g0, g1, gz, gz, gz8, 0);
#else
        __builtin_amdgcn_tensor_load_to_lds(g0, g1, gz, gz, 0);
#endif
      }
#else
      // Fallback: per-lane async copies, manually unrolled (<=2 issues).
      unsigned l32 = (unsigned)(size_t)lbase + t * 16;
      async_b128_to_lds(l32, gbase + t * 16);
      if (NT == 8) {
        async_b128_to_lds(l32 + 4096, gbase + t * 16 + 4096);
      } else if (t < (NT * 1024 - 4096) / 16) {
        async_b128_to_lds(l32 + 4096, gbase + t * 16 + 4096);
      }
#endif
      // prefetch next K panel into GL2
      if ((k0 + BK) < C_ && t == 0) {
        __builtin_prefetch((const void*)(gbase + (size_t)NTtot * 512 * 2), 0, 0);
      }
    }
    // ------- A tile: fp32 float4 loads -> bf16 fragment-order scatter ----
    if (MODE == 2) {
      // row-major: Asrc[(b*LQ+l)*C + k], float4 along k
      #pragma unroll
      for (int i = 0; i < 4; ++i) {
        int idx = t + i * 256;              // 0..1023
        int kv  = idx & 7;                  // k4 group
        int ll  = idx >> 3;                 // row 0..127
        const float4 v4 = *reinterpret_cast<const float4*>(
            &Asrc[(size_t)(rowBlock + ll) * C_ + k0 + kv * 4]);
        float fv[4] = {v4.x, v4.y, v4.z, v4.w};
        #pragma unroll
        for (int j = 0; j < 4; ++j) {
          int k = kv * 4 + j, lh, el;
          frag_slot(k, lh, el);
          ((__bf16*)&AfragLDS[ll >> 4][lh * 16 + (ll & 15)])[el] = f2bf(fv[j]);
        }
      }
    } else {
      // transposed: Asrc[b*C*LQ + k*LQ + l], float4 along l
      #pragma unroll
      for (int i = 0; i < 4; ++i) {
        int idx = t + i * 256;              // 0..1023
        int lv  = idx & 31;                 // l4 group
        int kk  = idx >> 5;                 // 0..31
        const float4 v4 = *reinterpret_cast<const float4*>(
            &Asrc[(size_t)b * C_ * LQ + (size_t)(k0 + kk) * LQ + l0 + lv * 4]);
        int lh, el;
        frag_slot(kk, lh, el);
        float fv[4] = {v4.x, v4.y, v4.z, v4.w};
        #pragma unroll
        for (int j = 0; j < 4; ++j) {
          int row = lv * 4 + j;
          ((__bf16*)&AfragLDS[row >> 4][lh * 16 + (row & 15)])[el] = f2bf(fv[j]);
        }
      }
    }
    // All B DMA + DS A stores must be visible to every wave after barrier.
#if defined(USE_TDM)
    __builtin_amdgcn_s_wait_tensorcnt(0);
#else
    asm volatile("s_wait_asynccnt 0x0" ::: "memory");
#endif
    __syncthreads();

    // ------- fragments: contiguous 32B LDS reads, then WMMA chain --------
    v16bf afrag = AfragLDS[wave][lane];
    v16bf bfr[NT];
    #pragma unroll
    for (int nt = 0; nt < NT; ++nt) bfr[nt] = BfragLDS[nt][lane];
    #pragma unroll
    for (int nt = 0; nt < NT; ++nt) {
      acc[nt] = __builtin_amdgcn_wmma_f32_16x16x32_bf16(
          false, afrag, false, bfr[nt], (short)0, acc[nt], false, false);
    }
    __syncthreads();
  }

  // ---------------- epilogue ----------------
  // C/D layout: element r of v8f -> M = r + 8*laneHalf, N = laneMod.
  #pragma unroll
  for (int nt = 0; nt < NT; ++nt) {
    int n = colBlock + nt * 16 + laneMod;
    #pragma unroll
    for (int r = 0; r < 8; ++r) {
      int row = wave * 16 + r + laneHalf * 8;
      int l   = l0 + row;
      float v = acc[nt][r];
      if (MODE == 1) {
        v += (n < 64) ? bias[n] : bias2[n - 64];
        Out[((size_t)b * LQ + l) * 96 + n] = v;
      } else if (MODE == 0) {
        v += bias[n];
        Out[((size_t)b * LQ + l) * C_ + n] = v;
      } else {
        v += bias[n] + bias2[(size_t)b * C_ * LQ + (size_t)n * LQ + l];
        Out[(size_t)b * C_ * LQ + (size_t)n * LQ + l] = v;  // [B,C,Lq] store
      }
    }
  }
}

// One wave per (b, m, q); lane == channel d (D=32 matches wave32).
// value is [B, Lq, C] so the 32 channels of (position, head) are one
// contiguous 128B cacheline -> each corner gather is a single coalesced load.
__global__ __launch_bounds__(256) void deform_sample_kernel(
    const float* __restrict__ value,   // [B, Lq, C]
    const float* __restrict__ proj,    // [B, Lq, 96]  (64 offsets | 32 logits)
    float* __restrict__ out) {         // [B, Lq, C]
  const int wave = threadIdx.x >> 5;
  const int lane = threadIdx.x & 31;
  long task = (long)blockIdx.x * 8 + wave;          // B*M*Lq total tasks
  int q  = (int)(task % LQ);
  int bm = (int)(task / LQ);
  int m  = bm & (M_ - 1);
  int b  = bm >> 3;
  int y  = q >> 7;            // q / W_
  int x  = q & (W_ - 1);

  const float4* pr4 =
      reinterpret_cast<const float4*>(proj + ((size_t)b * LQ + q) * 96);
  float4 o0  = pr4[m * 2];          // offsets for this head: 8 floats
  float4 o1  = pr4[m * 2 + 1];
  float4 lgv = pr4[16 + m];         // 4 attention logits
  float off[8] = {o0.x, o0.y, o0.z, o0.w, o1.x, o1.y, o1.z, o1.w};
  float lg[4]  = {lgv.x, lgv.y, lgv.z, lgv.w};

  // softmax over P=4
  float mx = fmaxf(fmaxf(lg[0], lg[1]), fmaxf(lg[2], lg[3]));
  float e[4], s = 0.f;
  #pragma unroll
  for (int p = 0; p < 4; ++p) { e[p] = __expf(lg[p] - mx); s += e[p]; }
  float inv = 1.0f / s;

  // Reference's (B,M,Lq,2,P)->(B*M,Lq,P,2) reshape mixes coord & point axes:
  // flat j in [0,8): coord c = j>>2, point pp = j&3.
  float locn[8];
  #pragma unroll
  for (int j = 0; j < 8; ++j) {
    int c  = j >> 2;
    int pp = j & 3;
    float base = (c == 0 ? (float)x : (float)y) + 0.5f;
    float dim  = (c == 0 ? (float)W_ : (float)H_);
    locn[j] = (base + off[pp * 2 + c]) / dim;
  }

  const float* vbase = value + (size_t)b * LQ * C_ + m * D_ + lane;
  float acc = 0.f;
  #pragma unroll
  for (int p = 0; p < 4; ++p) {
    float gx = locn[2 * p]     * (float)W_ - 0.5f;
    float gy = locn[2 * p + 1] * (float)H_ - 0.5f;
    float x0f = floorf(gx), y0f = floorf(gy);
    float wx1 = gx - x0f, wy1 = gy - y0f;
    float aw = e[p] * inv;
    #pragma unroll
    for (int cy = 0; cy < 2; ++cy) {
      #pragma unroll
      for (int cx = 0; cx < 2; ++cx) {
        float xf = x0f + cx, yf = y0f + cy;
        bool valid = (xf >= 0.f) && (xf <= (float)(W_ - 1)) &&
                     (yf >= 0.f) && (yf <= (float)(H_ - 1));
        if (valid) {
          float w = (cx ? wx1 : 1.f - wx1) * (cy ? wy1 : 1.f - wy1);
          int xi = (int)xf, yi = (int)yf;
          acc += aw * w * vbase[(size_t)(yi * W_ + xi) * C_];
        }
      }
    }
  }
  out[((size_t)b * LQ + q) * C_ + m * D_ + lane] = acc;
}

extern "C" void kernel_launch(void* const* d_in, const int* in_sizes, int n_in,
                              void* d_out, int out_size, void* d_ws,
                              size_t ws_size, hipStream_t stream) {
  (void)in_sizes; (void)n_in; (void)out_size; (void)ws_size;
  const float* ego    = (const float*)d_in[0];
  const float* collab = (const float*)d_in[1];
  const float* W_off  = (const float*)d_in[2];
  const float* b_off  = (const float*)d_in[3];
  const float* W_attn = (const float*)d_in[4];
  const float* b_attn = (const float*)d_in[5];
  const float* W_val  = (const float*)d_in[6];
  const float* b_val  = (const float*)d_in[7];
  const float* W_out  = (const float*)d_in[8];
  const float* b_out  = (const float*)d_in[9];
  float* out = (float*)d_out;

  float* value_ws = (float*)d_ws;                          // B*Lq*C  (64 MB)
  float* proj_ws  = value_ws + (size_t)B_ * LQ * C_;       // B*Lq*96 (24 MB)
  float* samp_ws  = proj_ws  + (size_t)B_ * LQ * 96;       // B*Lq*C  (64 MB)
  __bf16* wval_p  = (__bf16*)(samp_ws + (size_t)B_ * LQ * C_);  // 128 KB
  __bf16* wout_p  = wval_p + (size_t)C_ * C_;                   // 128 KB
  __bf16* wproj_p = wout_p + (size_t)C_ * C_;                   // 48 KB

  dim3 blk(256);
  // K0: pack weights into bf16 fragment-order panels (one-time per call)
  pack_weights_kernel<<<dim3(C_), blk, 0, stream>>>(W_val, nullptr, C_, C_, wval_p);
  pack_weights_kernel<<<dim3(C_), blk, 0, stream>>>(W_out, nullptr, C_, C_, wout_p);
  pack_weights_kernel<<<dim3(96), blk, 0, stream>>>(W_off, W_attn, 96, 64, wproj_p);

  // K1: value projection GEMM (bf16 WMMA, f32 acc)
  gemm_wmma_kernel<0, 128><<<dim3((B_ * LQ) / 128, C_ / 128), blk, 0, stream>>>(
      collab, wval_p, b_val, nullptr, value_ws);
  // K2: fused offset+attention projection GEMM
  gemm_wmma_kernel<1, 96><<<dim3((B_ * LQ) / 128, 1), blk, 0, stream>>>(
      ego, wproj_p, b_off, b_attn, proj_ws);
  // K3: softmax + deformable bilinear gather
  deform_sample_kernel<<<dim3((B_ * M_ * LQ) / 8), blk, 0, stream>>>(
      value_ws, proj_ws, samp_ws);
  // K4: output projection GEMM + bias + residual, transposed store
  gemm_wmma_kernel<2, 128><<<dim3((B_ * LQ) / 128, C_ / 128), blk, 0, stream>>>(
      samp_ws, wout_p, b_out, ego, out);
}
